// Involution_57982058496731
// MI455X (gfx1250) — matrix operand order repaired and checked by reading, not compiled, not run on hardware
//
#include <hip/hip_runtime.h>
#include <hip/hip_bf16.h>

// ---- problem constants (from reference) ----
#define C_IN   256          // input channels
#define RCH    128          // reduced channels
#define GRP    64           // groups
#define GS     4            // group size
#define KK     49           // K*K taps
#define OCH    3136         // patch_conv out channels = KK*GRP
#define HDIM   56
#define WDIM   56
#define HW     3136         // H*W
#define BATCH  8
#define NT     16           // pixel tile (columns of the WMMA C tile)
#define NTILES (HW / NT)    // 196 pixel tiles per image (exact)
#define YPAD   (RCH + 8)    // padded LDS row (halves): 272 B -> 4-bank rotate/col

typedef __attribute__((ext_vector_type(2)))  float   v2f;
typedef __attribute__((ext_vector_type(8)))  float   v8f;
typedef __attribute__((ext_vector_type(8)))  __bf16  v8bf;
typedef __attribute__((ext_vector_type(16))) __bf16  v16bf;

// fp32 -> bf16 with round-to-nearest-even (pure integer ops)
__device__ __forceinline__ unsigned short f2bf(float f) {
    unsigned u = __float_as_uint(f);
    return (unsigned short)((u + 0x7fffu + ((u >> 16) & 1u)) >> 16);
}
__device__ __forceinline__ float bf2f(unsigned short h) {
    return __uint_as_float((unsigned)h << 16);
}

__device__ __forceinline__ v16bf mk16(v8bf lo, v8bf hi) {
    return __builtin_shufflevector(lo, hi, 0, 1, 2, 3, 4, 5, 6, 7,
                                           8, 9, 10, 11, 12, 13, 14, 15);
}

// ---- gfx1250 async Global->LDS copy (inline asm; ASYNCcnt-tracked) ----
// LDS destination is a 32-bit offset within the wave's LDS allocation:
// take ptrtoint of the addrspace(3) view of the shared-memory pointer.
__device__ __forceinline__ unsigned lds_off32(const void* p) {
    return (unsigned)(unsigned long long)
           (const __attribute__((address_space(3))) char*)p;
}
__device__ __forceinline__ void async_copy_b128(void* lds_dst, const void* gsrc) {
    unsigned dst = lds_off32(lds_dst);
    asm volatile("global_load_async_to_lds_b128 %0, %1, off"
                 :: "v"(dst), "v"(gsrc) : "memory");
}
__device__ __forceinline__ void async_wait0() {
    asm volatile("s_wait_asynccnt 0x0" ::: "memory");
}

// =====================================================================
// Kernel 0: split w2 into bf16 (hi, lo) pair arrays, row-major OC x RCH.
// =====================================================================
__global__ __launch_bounds__(256) void split_w2_kernel(
    const float* __restrict__ w2,
    unsigned short* __restrict__ w2h, unsigned short* __restrict__ w2l)
{
    int i = blockIdx.x * 256 + threadIdx.x;
    if (i < OCH * RCH) {
        float f = w2[i];
        unsigned short h = f2bf(f);
        w2h[i] = h;
        w2l[i] = f2bf(f - bf2f(h));
    }
}

// =====================================================================
// Kernel 1: y[b, m, p] = sum_k w1[m,k] * x[b,k,p] + b1[m]
// M=128 (8 waves x 16 rows), K=256, N=16-pixel tile. Exact fp32 WMMA:
// this stage is memory-bound (~38 MB HBM), so the K=4 fp32 matrix op is
// free here and keeps GEMM1 bit-faithful. The x tile is staged with the
// gfx1250 async Global->LDS path (no VGPR round-trip).
// =====================================================================
__global__ __launch_bounds__(256) void gemm1_kernel(
    const float* __restrict__ x, const float* __restrict__ w1,
    const float* __restrict__ b1, float* __restrict__ y)
{
    __shared__ float lds_x[C_IN * NT];            // 16 KB: B-matrix tile (K x N)

    const int tile = blockIdx.x;                  // 0 .. BATCH*NTILES-1
    const int b    = tile / NTILES;
    const int p0   = (tile % NTILES) * NT;
    const int tid  = threadIdx.x;

    const float* xb = x + (size_t)b * C_IN * HW;
    // async-stage x tile: 1024 x 16B chunks, all 16B-aligned
#pragma unroll
    for (int i = tid; i < (C_IN * NT) / 4; i += 256) {
        int k = i >> 2, c4 = (i & 3) * 4;
        async_copy_b128(&lds_x[k * NT + c4], xb + (size_t)k * HW + p0 + c4);
    }
    async_wait0();
    __syncthreads();

    const int wave = tid >> 5;
    const int lane = tid & 31;
    const int ln15 = lane & 15;
    const int koff = (lane >> 4) * 2;             // f32 A/B frag K offset
    const int m0   = wave * 16;

    v8f acc = {};
    const float* arow = w1 + (size_t)(m0 + ln15) * C_IN + koff;
    for (int k0 = 0; k0 < C_IN; k0 += 4) {
        v2f a, bf;
        a.x  = arow[k0];
        a.y  = arow[k0 + 1];
        bf.x = lds_x[(k0 + koff)     * NT + ln15];
        bf.y = lds_x[(k0 + koff + 1) * NT + ln15];
        acc = __builtin_amdgcn_wmma_f32_16x16x4_f32(
                  false, a, false, bf, (short)0, acc, false, false);
    }

    const int col   = ln15;
    const int rbase = 8 * (lane >> 4);
    float* yb = y + (size_t)b * RCH * HW + p0 + col;
#pragma unroll
    for (int v = 0; v < 8; ++v) {
        int m = m0 + rbase + v;
        yb[(size_t)m * HW] = acc[v] + b1[m];
    }
}

// =====================================================================
// Kernel 2: wt = w2·y + b2 (M=3136, K=128, N=16) via bf16 pair-split
// WMMA (a_hi·b_hi + a_hi·b_lo + a_lo·b_hi in fp32 accumulators), fused
// with the involution: each lane's 8 consecutive oc rows share a group
// (crossing at most one boundary), so the 4 group-channel partial sums
// are kept in registers and flushed to the LDS accumulator once per
// group (4-8 ds_add_f32 per tile instead of 32).
// =====================================================================
__global__ __launch_bounds__(256) void gemm2_inv_kernel(
    const float* __restrict__ x, const float* __restrict__ y,
    const unsigned short* __restrict__ w2h,
    const unsigned short* __restrict__ w2l,
    const float* __restrict__ b2, float* __restrict__ out)
{
    __shared__ unsigned short lds_yh[NT * YPAD];  // 4.25 KB, transposed [col][k]
    __shared__ unsigned short lds_yl[NT * YPAD];  // 4.25 KB
    __shared__ float          lds_out[C_IN * NT]; // 16 KB output accumulator

    const int tile = blockIdx.x;
    const int b    = tile / NTILES;
    const int p0   = (tile % NTILES) * NT;
    const int tid  = threadIdx.x;

    // Stage y tile: read fp32, split into bf16 hi/lo, store K-contiguous
    // per column so B-fragments are single 16B ds reads.
    const float* ybase = y + (size_t)b * RCH * HW;
    for (int i = tid; i < RCH * NT; i += 256) {
        int k = i >> 4, col = i & 15;
        float f = ybase[(size_t)k * HW + p0 + col];
        unsigned short h = f2bf(f);
        lds_yh[col * YPAD + k] = h;
        lds_yl[col * YPAD + k] = f2bf(f - bf2f(h));
    }
    for (int i = tid; i < C_IN * NT; i += 256) lds_out[i] = 0.0f;
    __syncthreads();

    const int wave  = tid >> 5;
    const int lane  = tid & 31;
    const int ln15  = lane & 15;
    const int col   = ln15;
    const int rbase = 8 * (lane >> 4);
    const int kbA   = 16 * (lane >> 4);   // bf16 A layout: K = kbA+0..15
    const int kbB   = 16 * (lane >> 4);   // bf16 B layout: K = kbB+0..15

    const int p  = p0 + col;
    const int hh = p / WDIM;
    const int ww = p % WDIM;
    const float* xb = x + (size_t)b * C_IN * HW;

    for (int mt = wave; mt < OCH / 16; mt += 8) {
        const int m0     = mt * 16;
        const int rowoff = (m0 + ln15) * RCH;     // in halves

        v8f acc = {};
#pragma unroll
        for (int kc = 0; kc < RCH; kc += 32) {
            v16bf Ah = mk16(*(const v8bf*)(w2h + rowoff + kc + kbA),
                            *(const v8bf*)(w2h + rowoff + kc + kbA + 8));
            v16bf Al = mk16(*(const v8bf*)(w2l + rowoff + kc + kbA),
                            *(const v8bf*)(w2l + rowoff + kc + kbA + 8));
            const unsigned short* byh = lds_yh + col * YPAD + kc + kbB;
            const unsigned short* byl = lds_yl + col * YPAD + kc + kbB;
            v16bf Bh = mk16(*(const v8bf*)(byh), *(const v8bf*)(byh + 8));
            v16bf Bl = mk16(*(const v8bf*)(byl), *(const v8bf*)(byl + 8));

            acc = __builtin_amdgcn_wmma_f32_16x16x32_bf16(
                      false, Ah, false, Bh, (short)0, acc, false, false);
            acc = __builtin_amdgcn_wmma_f32_16x16x32_bf16(
                      false, Ah, false, Bl, (short)0, acc, false, false);
            acc = __builtin_amdgcn_wmma_f32_16x16x32_bf16(
                      false, Al, false, Bh, (short)0, acc, false, false);
        }

        // Fused involution epilogue with per-group register accumulation.
        float s0 = 0.f, s1 = 0.f, s2 = 0.f, s3 = 0.f;
        int gacc = (m0 + rbase) / KK;
#pragma unroll
        for (int v = 0; v < 8; ++v) {
            const int oc = m0 + rbase + v;
            const int g  = oc / KK;               // group
            const int q  = oc - g * KK;           // tap index 0..48
            if (g != gacc) {                      // crossed a group boundary
                const int base = (gacc * GS) * NT + col;
                atomicAdd(&lds_out[base         ], s0);
                atomicAdd(&lds_out[base +     NT], s1);
                atomicAdd(&lds_out[base + 2 * NT], s2);
                atomicAdd(&lds_out[base + 3 * NT], s3);
                s0 = s1 = s2 = s3 = 0.f;
                gacc = g;
            }
            const float wt = acc[v] + b2[oc];
            const int ki = q / 7 - 3;             // pad = 3
            const int kj = (q - (q / 7) * 7) - 3;
            const int xh = hh + ki;
            const int xw = ww + kj;
            if ((xh >= 0) & (xh < HDIM) & (xw >= 0) & (xw < WDIM)) {
                const float* xpix = xb + (size_t)(g * GS) * HW
                                       + (size_t)xh * WDIM + xw;
                s0 += wt * xpix[0];
                s1 += wt * xpix[HW];
                s2 += wt * xpix[2 * HW];
                s3 += wt * xpix[3 * HW];
            }
        }
        {
            const int base = (gacc * GS) * NT + col;
            atomicAdd(&lds_out[base         ], s0);
            atomicAdd(&lds_out[base +     NT], s1);
            atomicAdd(&lds_out[base + 2 * NT], s2);
            atomicAdd(&lds_out[base + 3 * NT], s3);
        }
    }
    __syncthreads();

    float* ob = out + (size_t)b * C_IN * HW + p0;
    for (int i = tid; i < C_IN * NT; i += 256) {
        int ch = i / NT, c2 = i % NT;
        ob[(size_t)ch * HW + c2] = lds_out[i];
    }
}

// =====================================================================
// Launch
// =====================================================================
extern "C" void kernel_launch(void* const* d_in, const int* in_sizes, int n_in,
                              void* d_out, int out_size, void* d_ws, size_t ws_size,
                              hipStream_t stream)
{
    const float* x  = (const float*)d_in[0];
    const float* w1 = (const float*)d_in[1];
    const float* b1 = (const float*)d_in[2];
    const float* w2 = (const float*)d_in[3];
    const float* b2 = (const float*)d_in[4];
    float* out = (float*)d_out;

    // workspace layout: y (fp32) | w2_hi (bf16) | w2_lo (bf16)
    const size_t Y_ELEMS = (size_t)BATCH * RCH * HW;
    float*          ybuf = (float*)d_ws;
    unsigned short* w2h  = (unsigned short*)(ybuf + Y_ELEMS);
    unsigned short* w2l  = w2h + (size_t)OCH * RCH;

    const int nblk = BATCH * NTILES;   // 1568 workgroups of 8 wave32s
    split_w2_kernel<<<(OCH * RCH + 255) / 256, 256, 0, stream>>>(w2, w2h, w2l);
    gemm1_kernel<<<nblk, 256, 0, stream>>>(x, w1, b1, ybuf);
    gemm2_inv_kernel<<<nblk, 256, 0, stream>>>(x, ybuf, w2h, w2l, b2, out);
}
